// RelPN_55018531062328
// MI455X (gfx1250) — compile-verified, matrix operand order checked by reading, not run on hardware
//
#include <hip/hip_runtime.h>

// ---------------------------------------------------------------------------
// RelPN for MI455X (gfx1250, wave32, WMMA):
//   K0a/K0b: pack W1/W2 into WMMA B-fragment lane order (bf16)
//   K1: fused 2-layer MLP via v_wmma_f32_16x16x32_bf16; emits subj/obj as
//       pre-packed bf16 WMMA fragments (A/B layouts are identical here)
//   K2: raw dots subj_tile @ obj^T via WMMA (all operands = 32B/lane vector
//       loads), fused exact per-tile top-64 in LDS (sigmoid deferred)
//   K3: per-batch merge (async global->LDS staging), sigmoid on 64 winners
// ---------------------------------------------------------------------------

typedef __attribute__((ext_vector_type(16))) __bf16 v16bf;
typedef __attribute__((ext_vector_type(8)))  __bf16 v8bf;
typedef __attribute__((ext_vector_type(8)))  float  v8f;

static constexpr int B    = 32;
static constexpr int N    = 1000;
static constexpr int NT   = 63;     // 16-row tiles over N (63*16 = 1008)
static constexpr int CIN  = 1601;
static constexpr int KPAD = 1632;   // 51 * 32
static constexpr int KIT  = 51;
static constexpr int HID  = 64;
static constexpr int TAKE = 64;
static constexpr int FRAG = 32 * 16;   // bf16 elements per packed fragment

#define NEG_INF (-__builtin_inff())

// Per-lane K index for the 16-bit 16x32 fragment layout (wave32):
// element e of lane l: K = (e/8)*16 + (l/16)*8 + (e%8)
__device__ __forceinline__ int kmap(int e, int khalf) {
    return ((e >> 3) << 4) + (khalf << 3) + (e & 7);
}

__device__ __forceinline__ v8f wmma_bf16(const v16bf& a, const v16bf& b, const v8f& c) {
    return __builtin_amdgcn_wmma_f32_16x16x32_bf16(false, a, false, b, (short)0, c, false, false);
}

// ---------------------------------------------------------------------------
// K0a: pack layer-1 weights [CIN x 64] -> fragments [2][4][KIT][32][16] bf16.
// One (fragment, lane) per thread.
// ---------------------------------------------------------------------------
__global__ __launch_bounds__(256) void pack_w1_kernel(
    const float* __restrict__ Ws1, const float* __restrict__ Wo1,
    __bf16* __restrict__ pW1)
{
    const int t = blockIdx.x * 256 + threadIdx.x;
    const int total = 2 * 4 * KIT * 32;
    if (t >= total) return;
    const int lane  = t & 31;
    const int f     = t >> 5;              // ((which*4 + wn)*KIT + kt)
    const int kt    = f % KIT;
    const int wn    = (f / KIT) & 3;
    const int which = f / (KIT * 4);
    const float* W  = which ? Wo1 : Ws1;
    const int mcol  = lane & 15, khalf = lane >> 4;
    const int ncol  = wn * 16 + mcol;
    v16bf o;
#pragma unroll
    for (int e = 0; e < 16; ++e) {
        const int kg = kt * 32 + kmap(e, khalf);
        o[e] = (kg < CIN) ? (__bf16)W[kg * HID + ncol] : (__bf16)0.0f;
    }
    *(v16bf*)(pW1 + (size_t)t * 16) = o;
}

// K0b: pack layer-2 weights [64 x 64] -> fragments [2][4][2][32][16] bf16.
__global__ __launch_bounds__(256) void pack_w2_kernel(
    const float* __restrict__ Ws2, const float* __restrict__ Wo2,
    __bf16* __restrict__ pW2)
{
    const int t = blockIdx.x * 256 + threadIdx.x;
    const int total = 2 * 4 * 2 * 32;
    if (t >= total) return;
    const int lane  = t & 31;
    const int f     = t >> 5;              // ((which*4 + wn)*2 + kt)
    const int kt    = f & 1;
    const int wn    = (f >> 1) & 3;
    const int which = f >> 3;
    const float* W  = which ? Wo2 : Ws2;
    const int mcol  = lane & 15, khalf = lane >> 4;
    const int ncol  = wn * 16 + mcol;
    v16bf o;
#pragma unroll
    for (int e = 0; e < 16; ++e) {
        const int kg = kt * 32 + kmap(e, khalf);
        o[e] = (__bf16)W[kg * HID + ncol];
    }
    *(v16bf*)(pW2 + (size_t)t * 16) = o;
}

// ---------------------------------------------------------------------------
// K1: fused 2-layer MLP for one 16-row tile, emitting packed fragments.
// grid = (32, 63), block = 256 (8 waves). Waves 0-3: subj cols, 4-7: obj cols.
// ---------------------------------------------------------------------------
__global__ __launch_bounds__(256) void relpn_mlp_kernel(
    const float* __restrict__ x,                                    // [B,N,CIN]
    const __bf16* __restrict__ pW1, const __bf16* __restrict__ pW2, // packed
    const float* __restrict__ bs1, const float* __restrict__ bs2,
    const float* __restrict__ bo1, const float* __restrict__ bo2,
    __bf16* __restrict__ subjP, __bf16* __restrict__ objP)          // [B,NT,2,32,16]
{
    __shared__ __align__(16) __bf16 xs[16 * KPAD];   // 52 KB bf16 x-tile
    __shared__ __align__(16) float  hs[2][16 * HID]; //  8 KB activations

    const int b    = blockIdx.x;
    const int tile = blockIdx.y;
    const int tid  = threadIdx.x;
    const int row0 = tile * 16;

    // Stage + convert x tile (f32 -> bf16), zero-padded to K=1632 / rows >= N.
    const float* xb = x + (size_t)b * N * CIN;
    for (int idx = tid; idx < 16 * KPAD; idx += 256) {
        int r = idx / KPAD, c = idx - r * KPAD;
        int grow = row0 + r;
        float v = (c < CIN && grow < N) ? xb[(size_t)grow * CIN + c] : 0.0f;
        xs[idx] = (__bf16)v;
    }
    __syncthreads();

    const int wave  = tid >> 5;
    const int lane  = tid & 31;
    const int wn    = wave & 3;
    const int isObj = wave >> 2;
    const int mcol  = lane & 15;
    const int khalf = lane >> 4;
    const int ncol  = wn * 16 + mcol;

    const __bf16* W1 = pW1 + (size_t)isObj * 4 * KIT * FRAG;
    const __bf16* W2 = pW2 + (size_t)isObj * 4 * 2 * FRAG;
    const float*  b1 = isObj ? bo1 : bs1;
    const float*  b2 = isObj ? bo2 : bs2;

    // ---- layer 1: x(16x1601) @ W1 -> 16x16 tile per wave ----
    v8f acc = {};
    const v8bf* xrow = (const v8bf*)(xs + mcol * KPAD);  // 8-elem groups of 16B
    for (int kt = 0; kt < KIT; ++kt) {
        // A: two contiguous 16B LDS loads selected by lane half.
        const v8bf lo = xrow[kt * 4 + khalf];
        const v8bf hi = xrow[kt * 4 + 2 + khalf];
        const v16bf a = __builtin_shufflevector(lo, hi,
            0, 1, 2, 3, 4, 5, 6, 7, 8, 9, 10, 11, 12, 13, 14, 15);
        // B: one contiguous 32B/lane global load from packed weights.
        const v16bf bb = *(const v16bf*)(W1 + ((size_t)(wn * KIT + kt) * 32 + lane) * 16);
        acc = wmma_bf16(a, bb, acc);
    }
    const float bias1 = b1[ncol];
#pragma unroll
    for (int r = 0; r < 8; ++r) {
        float h = acc[r] + bias1;
        h = h > 0.0f ? h : 0.0f;                 // ReLU
        hs[isObj][(r + (khalf << 3)) * HID + ncol] = h;
    }
    __syncthreads();

    // ---- layer 2: h(16x64) @ W2(64x64), K = 64 -> 2 WMMAs ----
    v8f acc2 = {};
#pragma unroll
    for (int kt = 0; kt < 2; ++kt) {
        v16bf a;
#pragma unroll
        for (int e = 0; e < 16; ++e)
            a[e] = (__bf16)hs[isObj][mcol * HID + kt * 32 + kmap(e, khalf)];
        const v16bf bb = *(const v16bf*)(W2 + ((size_t)(wn * 2 + kt) * 32 + lane) * 16);
        acc2 = wmma_bf16(a, bb, acc2);
    }
    const float bias2 = b2[ncol];
    __syncthreads();                 // all layer-2 reads of hs done
#pragma unroll
    for (int r = 0; r < 8; ++r)
        hs[isObj][(r + (khalf << 3)) * HID + ncol] = acc2[r] + bias2;
    __syncthreads();

    // ---- pack subj/obj as bf16 fragments (A layout == B layout here) ----
    if (wave < 4) {
        const int src = wave >> 1;   // 0 = subj, 1 = obj
        const int kt  = wave & 1;
        v16bf o;
#pragma unroll
        for (int e = 0; e < 16; ++e)
            o[e] = (__bf16)hs[src][mcol * HID + kt * 32 + kmap(e, khalf)];
        __bf16* dst = (src ? objP : subjP) +
                      (((size_t)(b * NT + tile) * 2 + kt) * 32 + lane) * 16;
        *(v16bf*)dst = o;
    }
}

// ---------------------------------------------------------------------------
// K2: raw dots for 16 subject rows x all objects; exact local top-64 in LDS.
// grid = (32, 63), block = 256 (8 waves share j-tiles). Sigmoid deferred.
// ---------------------------------------------------------------------------
__global__ __launch_bounds__(256) void relpn_score_topk_kernel(
    const __bf16* __restrict__ subjP, const __bf16* __restrict__ objP,
    float* __restrict__ cand_val, int* __restrict__ cand_idx)   // [B, 63*64]
{
    __shared__ __align__(16) float sc[16 * N];    // 64 KB raw dot scores
    __shared__ float redv[8];
    __shared__ int   redp[8];

    const int b     = blockIdx.x;
    const int itile = blockIdx.y;
    const int tid   = threadIdx.x;

    for (int i = tid; i < 16 * N; i += 256) sc[i] = NEG_INF;
    __syncthreads();

    const int wave  = tid >> 5;
    const int lane  = tid & 31;
    const int mcol  = lane & 15;
    const int khalf = lane >> 4;

    // A fragments (subject rows): two 32B/lane loads, reused for all j-tiles.
    const __bf16* sfrag = subjP + ((size_t)(b * NT + itile) * 2) * FRAG + lane * 16;
    const v16bf a0 = *(const v16bf*)(sfrag);
    const v16bf a1 = *(const v16bf*)(sfrag + FRAG);

    const __bf16* ofragB = objP + (size_t)b * NT * 2 * FRAG + lane * 16;
    for (int jt = wave; jt < NT; jt += 8) {
        const __bf16* of = ofragB + (size_t)jt * 2 * FRAG;
        if (jt + 8 < NT)
            __builtin_prefetch(of + (size_t)8 * 2 * FRAG, 0, 0);  // global_prefetch_b8
        const v16bf b0 = *(const v16bf*)(of);
        const v16bf b1 = *(const v16bf*)(of + FRAG);
        v8f c = {};
        c = wmma_bf16(a0, b0, c);
        c = wmma_bf16(a1, b1, c);
        const int jg = jt * 16 + mcol;
#pragma unroll
        for (int r = 0; r < 8; ++r) {
            const int il = r + (khalf << 3);
            float s = c[r];                               // raw dot (monotonic)
            if (itile * 16 + il >= N) s = NEG_INF;        // masked rows
            if (jg < N) sc[il * N + jg] = s;
        }
    }
    __syncthreads();

    // Exact local top-64 by iterative max extraction (float4 scans).
    float* cv = cand_val + ((size_t)b * NT + itile) * TAKE;
    int*   ci = cand_idx + ((size_t)b * NT + itile) * TAKE;
    const float4* sc4 = (const float4*)sc;                // 4000 entries
    for (int k = 0; k < TAKE; ++k) {
        float best = NEG_INF; int bp = 0;
        for (int q = tid; q < 4000; q += 256) {
            const float4 w = sc4[q];
            const int p = q << 2;
            if (w.x > best) { best = w.x; bp = p; }
            if (w.y > best) { best = w.y; bp = p + 1; }
            if (w.z > best) { best = w.z; bp = p + 2; }
            if (w.w > best) { best = w.w; bp = p + 3; }
        }
#pragma unroll
        for (int off = 16; off > 0; off >>= 1) {          // wave32 reduce
            const float ov = __shfl_xor(best, off, 32);
            const int   op = __shfl_xor(bp,   off, 32);
            if (ov > best) { best = ov; bp = op; }
        }
        if (lane == 0) { redv[wave] = best; redp[wave] = bp; }
        __syncthreads();
        if (tid == 0) {
            float gb = redv[0]; int gp = redp[0];
#pragma unroll
            for (int w = 1; w < 8; ++w)
                if (redv[w] > gb) { gb = redv[w]; gp = redp[w]; }
            const int il = gp / N, jg = gp - il * N;
            cv[k] = gb;
            ci[k] = (itile * 16 + il) * N + jg;           // flat subj*N + obj
            sc[gp] = NEG_INF;                             // remove winner
        }
        __syncthreads();
    }
}

// ---------------------------------------------------------------------------
// K3: per-batch merge of 63*64 = 4032 candidates -> global top-64.
// Candidate staging uses the CDNA5 async global->LDS path. grid = 32.
// ---------------------------------------------------------------------------
__global__ __launch_bounds__(256) void relpn_merge_kernel(
    const float* __restrict__ cand_val, const int* __restrict__ cand_idx,
    int* __restrict__ out_pairs, float* __restrict__ out_scores)
{
    __shared__ __align__(16) float v[4096];
    __shared__ __align__(16) int   ix[4096];
    __shared__ float redv[256];
    __shared__ int   redp[256];

    const int b   = blockIdx.x;
    const int tid = threadIdx.x;
    const float* cv = cand_val + (size_t)b * NT * TAKE;
    const int*   ci = cand_idx + (size_t)b * NT * TAKE;

    // Async copy 4032 floats + 4032 ints into LDS (16B per transfer).
    for (int p = tid * 4; p < NT * TAKE; p += 1024) {
        const unsigned lv = (unsigned)(size_t)&v[p];
        const unsigned li = (unsigned)(size_t)&ix[p];
        const unsigned long long gv = (unsigned long long)(const void*)&cv[p];
        const unsigned long long gi = (unsigned long long)(const void*)&ci[p];
        asm volatile("global_load_async_to_lds_b128 %0, %1, off"
                     :: "v"(lv), "v"(gv) : "memory");
        asm volatile("global_load_async_to_lds_b128 %0, %1, off"
                     :: "v"(li), "v"(gi) : "memory");
    }
    asm volatile("s_wait_asynccnt 0x0" ::: "memory");
    __syncthreads();
    if (tid < 4096 - NT * TAKE) {                 // pad 4032..4095
        v[NT * TAKE + tid] = NEG_INF;
        ix[NT * TAKE + tid] = 0;
    }
    __syncthreads();

    const float4* v4 = (const float4*)v;          // 1024 entries
    for (int k = 0; k < TAKE; ++k) {
        float best = NEG_INF; int bp = 0;
        for (int q = tid; q < 1024; q += 256) {
            const float4 w = v4[q];
            const int p = q << 2;
            if (w.x > best) { best = w.x; bp = p; }
            if (w.y > best) { best = w.y; bp = p + 1; }
            if (w.z > best) { best = w.z; bp = p + 2; }
            if (w.w > best) { best = w.w; bp = p + 3; }
        }
        redv[tid] = best; redp[tid] = bp;
        __syncthreads();
#pragma unroll
        for (int s = 128; s > 0; s >>= 1) {
            if (tid < s && redv[tid + s] > redv[tid]) {
                redv[tid] = redv[tid + s]; redp[tid] = redp[tid + s];
            }
            __syncthreads();
        }
        if (tid == 0) {
            const int p   = redp[0];
            const int idx = ix[p];
            out_pairs[b * (TAKE * 2) + 2 * k]     = idx / N;   // subject
            out_pairs[b * (TAKE * 2) + 2 * k + 1] = idx % N;   // object
            out_scores[b * TAKE + k] = 1.0f / (1.0f + __expf(-redv[0]));  // sigmoid
            v[p] = NEG_INF;
        }
        __syncthreads();
    }
}

// ---------------------------------------------------------------------------
extern "C" void kernel_launch(void* const* d_in, const int* in_sizes, int n_in,
                              void* d_out, int out_size, void* d_ws, size_t ws_size,
                              hipStream_t stream) {
    const float* x   = (const float*)d_in[0];
    // d_in[1] = proposals (unused by the reference output)
    const float* Ws1 = (const float*)d_in[2];
    const float* bs1 = (const float*)d_in[3];
    const float* Ws2 = (const float*)d_in[4];
    const float* bs2 = (const float*)d_in[5];
    const float* Wo1 = (const float*)d_in[6];
    const float* bo1 = (const float*)d_in[7];
    const float* Wo2 = (const float*)d_in[8];
    const float* bo2 = (const float*)d_in[9];

    // Workspace carve-up (all chunks 32B-aligned; ~9.8 MB total).
    char* ws = (char*)d_ws;
    __bf16* subjP = (__bf16*)ws;                 // 32*63*2*512 bf16 = 4.13 MB
    ws += (size_t)B * NT * 2 * FRAG * sizeof(__bf16);
    __bf16* objP = (__bf16*)ws;
    ws += (size_t)B * NT * 2 * FRAG * sizeof(__bf16);
    __bf16* pW1 = (__bf16*)ws;                   // 2*4*51*512 bf16
    ws += (size_t)2 * 4 * KIT * FRAG * sizeof(__bf16);
    __bf16* pW2 = (__bf16*)ws;                   // 2*4*2*512 bf16
    ws += (size_t)2 * 4 * 2 * FRAG * sizeof(__bf16);
    float* cand_val = (float*)ws;                // 32*63*64 f32
    ws += (size_t)B * NT * TAKE * sizeof(float);
    int* cand_idx = (int*)ws;

    pack_w1_kernel<<<(2 * 4 * KIT * 32 + 255) / 256, 256, 0, stream>>>(Ws1, Wo1, pW1);
    pack_w2_kernel<<<1, 256, 0, stream>>>(Ws2, Wo2, pW2);

    dim3 g1(B, NT);
    relpn_mlp_kernel<<<g1, 256, 0, stream>>>(x, pW1, pW2, bs1, bs2, bo1, bo2,
                                             subjP, objP);

    dim3 g2(B, NT);
    relpn_score_topk_kernel<<<g2, 256, 0, stream>>>(subjP, objP, cand_val, cand_idx);

    // Output: pairs int32 [32,64,2] (4096 elems) then scores f32 [32,64] (2048).
    int*   out_pairs  = (int*)d_out;
    float* out_scores = (float*)d_out + (size_t)B * TAKE * 2;
    relpn_merge_kernel<<<B, 256, 0, stream>>>(cand_val, cand_idx, out_pairs, out_scores);
}
